// Model_61186104099559
// MI455X (gfx1250) — matrix operand order repaired and checked by reading, not compiled, not run on hardware
//
#include <hip/hip_runtime.h>
#include <hip/hip_bf16.h>
#include <cstdint>
#include <cstddef>

// ---------------------------------------------------------------------------
// CDNA5 (gfx1250) sparse-ResNet inference. All matmul work (convs as implicit
// GEMM + final linear) runs on v_wmma_f32_16x16x32_bf16 (wave32 layouts per
// CDNA5 ISA 7.12.2). B operand: transposed zero-padded bf16 weights
// -> two guard-free b128 loads per lane. A operand: two 8-wide K-runs per lane
// -> 4x float4 loads when Cin%8==0 (all layers but the 3-channel stem).
// ---------------------------------------------------------------------------

typedef __attribute__((ext_vector_type(16))) __bf16 v16bf;
typedef __attribute__((ext_vector_type(8)))  __bf16 v8bf;
typedef __attribute__((ext_vector_type(8)))  float  v8f;

#define BN_EPS 1e-4f

// ---------------- weight pack: [kh*kw*Cin, Cout] f32 -> [Coutp, Kp] bf16 ----
__global__ void k_pack_wT(const float* __restrict__ w, __bf16* __restrict__ out,
                          int K, int Cout, int Kp, int np) {
  int i = blockIdx.x * blockDim.x + threadIdx.x;
  if (i >= np) return;
  int k = i % Kp;
  int n = i / Kp;
  float v = 0.f;
  if (k < K && n < Cout) v = w[k * Cout + n];
  out[i] = (__bf16)v;
}

// ---------------- im2col decode table: tab[kk] = (r<<15)|(s<<10)|c ----------
__global__ void k_build_tab(int* __restrict__ tab, int kw, int Cin, int K) {
  int i = blockIdx.x * blockDim.x + threadIdx.x;
  if (i >= K) return;
  int c  = i % Cin;
  int rs = i / Cin;
  int s  = rs % kw;
  int r  = rs / kw;
  tab[i] = (r << 15) | (s << 10) | c;
}

// ---------------- pool_mask: any-active in window ----------------
__global__ void k_pool_mask(const float* __restrict__ mi, float* __restrict__ mo,
                            int B, int Hin, int Win, int Hout, int Wout, int size, int stride) {
  int i = blockIdx.x * blockDim.x + threadIdx.x;
  int n = B * Hout * Wout;
  if (i >= n) return;
  int ow = i % Wout; int r = i / Wout;
  int oh = r % Hout; int b = r / Hout;
  float mx = 0.f;
  for (int dy = 0; dy < size; ++dy)
    for (int dx = 0; dx < size; ++dx) {
      int ih = oh * stride + dy, iw = ow * stride + dx;
      if (ih < Hin && iw < Win) mx = fmaxf(mx, mi[(b * Hin + ih) * Win + iw]);
    }
  mo[i] = (mx > 0.f) ? 1.f : 0.f;
}

// ---------------- max-pool over ACTIVE sites ----------------
__global__ void k_maxpool_active(const float* __restrict__ h, const float* __restrict__ mi,
                                 const float* __restrict__ mo, float* __restrict__ out,
                                 int B, int Hin, int Win, int C, int Hout, int Wout,
                                 int size, int stride) {
  int i = blockIdx.x * blockDim.x + threadIdx.x;
  int n = B * Hout * Wout * C;
  if (i >= n) return;
  int c = i % C; int r = i / C;
  int ow = r % Wout; r /= Wout;
  int oh = r % Hout; int b = r / Hout;
  float best = -1e30f;
  for (int dy = 0; dy < size; ++dy)
    for (int dx = 0; dx < size; ++dx) {
      int ih = oh * stride + dy, iw = ow * stride + dx;
      if (ih < Hin && iw < Win) {
        int si = (b * Hin + ih) * Win + iw;
        float v = (mi[si] > 0.f) ? h[si * C + c] : -1e30f;
        best = fmaxf(best, v);
      }
    }
  int so = (b * Hout + oh) * Wout + ow;
  out[i] = (mo[so] > 0.f) ? best : 0.f;
}

// ---------------- count active sites ----------------
__global__ void k_mask_count(const float* __restrict__ m, float* __restrict__ cnt, int n) {
  __shared__ float sh[256];
  float s = 0.f;
  int step = gridDim.x * blockDim.x;
  for (int i = blockIdx.x * blockDim.x + threadIdx.x; i < n; i += step) s += m[i];
  sh[threadIdx.x] = s;
  __syncthreads();
  for (int o = 128; o > 0; o >>= 1) {
    if ((int)threadIdx.x < o) sh[threadIdx.x] += sh[threadIdx.x + o];
    __syncthreads();
  }
  if (threadIdx.x == 0) atomicAdd(cnt, sh[0]);
}

// ---------------- BN: per-channel sum & sumsq ----------------
// Inputs are exactly zero at inactive sites, so unmasked sums are correct.
__global__ void k_bn_reduce(const float* __restrict__ x, float* __restrict__ sums,
                            int C, int n) {
  extern __shared__ float sh[]; // 2*C
  for (int i = threadIdx.x; i < 2 * C; i += blockDim.x) sh[i] = 0.f;
  __syncthreads();
  int step = gridDim.x * blockDim.x;
  int i0 = blockIdx.x * blockDim.x + threadIdx.x;
  int c = i0 % C;
  int cstep = step % C;
  for (int i = i0; i < n; i += step) {
    float v = x[i];
    atomicAdd(&sh[c], v);
    atomicAdd(&sh[C + c], v * v);
    c += cstep; if (c >= C) c -= C;
  }
  __syncthreads();
  for (int i = threadIdx.x; i < 2 * C; i += blockDim.x)
    if (sh[i] != 0.f) atomicAdd(&sums[i], sh[i]);
}

// ---------------- BN apply + ReLU + mask zeroing ----------------
__global__ void k_bn_apply(const float* __restrict__ x, const float* __restrict__ m,
                           const float* __restrict__ sums, const float* __restrict__ cnt,
                           const float* __restrict__ g, const float* __restrict__ b,
                           float* __restrict__ out, int n, int C) {
  int i = blockIdx.x * blockDim.x + threadIdx.x;
  if (i >= n) return;
  int c = i % C;
  int site = i / C;
  float cn = fmaxf(cnt[0], 1.f);
  float mu = sums[c] / cn;
  float var = sums[C + c] / cn - mu * mu;
  float y = (x[i] - mu) * rsqrtf(var + BN_EPS) * g[c] + b[c];
  y = fmaxf(y, 0.f);
  out[i] = (m[site] > 0.f) ? y : 0.f;
}

// ---------------- implicit-GEMM conv / linear via WMMA bf16 ----------------
// One wave computes a 16 (sites) x 16 (out-channels) tile.
//   A (16x32 bf16): lane row M=lane&15; elements = two K-contiguous runs of 8
//     starting at k0+akb and k0+akb+16 (akb = 0 / 8 by lane half).
//   B (32x16 bf16): lane col N=lane&15; two K-contiguous runs of 8 starting at
//     k0+bkb and k0+bkb+16 (bkb = 0 / 8 by lane half) -> two b128 loads from [Coutp,Kp].
//   C/D (16x16 f32): col N=lane&15; VGPR r -> row (lane<16 ? r : r+8).
__global__ __launch_bounds__(32)
void k_conv_wmma(const float* __restrict__ in,      // [B,Hin,Win,Cin] f32
                 const __bf16* __restrict__ wt,     // [Coutp, Kp] bf16 (transposed, padded)
                 const int* __restrict__ tab,       // [K] im2col decode table
                 const float* __restrict__ mask,    // [B,Hout,Wout] or null
                 const float* __restrict__ resid,   // [B,Hout,Wout,Cout] or null
                 const float* __restrict__ bias,    // [Cout] or null
                 float* __restrict__ out,           // [B,Hout,Wout,Cout]
                 int M, int Hin, int Win, int Cin, int Cout,
                 int Wout, int HWo, int stride, int pad, int Kp, int K, int fastA) {
  __shared__ int tabLDS[2432];
  const int lane = threadIdx.x;
  for (int i = lane; i < K; i += 32) tabLDS[i] = tab[i];
  __syncthreads();

  const int mt = blockIdx.x;
  const int nt = blockIdx.y;

  // A mapping
  const int  gmA = mt * 16 + (lane & 15);
  const bool aok = gmA < M;
  const int  WinC = Win * Cin;
  int aBase = 0, ihB = 0, iwB = 0;
  if (aok) {
    int r = gmA;
    int ab = r / HWo; r -= ab * HWo;
    int aoh = r / Wout;
    int aow = r - aoh * Wout;
    ihB = aoh * stride - pad;
    iwB = aow * stride - pad;
    aBase = ab * (Hin * WinC);
  }
  const int akb = (lane >> 4) * 8;   // 0 or 8
  // B mapping (padded: always in-range)
  const int bn  = nt * 16 + (lane & 15);
  const int bkb = (lane >> 4) * 8;   // 0 or 8
  const __bf16* wrow = wt + bn * Kp + bkb;

  auto loadA = [&](int k0, v16bf& a) {
    if (fastA) {
      // two K-contiguous runs of 8, each within one (r,s) plane
#pragma unroll
      for (int h = 0; h < 2; ++h) {
        int kr = k0 + akb + h * 16;
        float buf[8] = {0.f, 0.f, 0.f, 0.f, 0.f, 0.f, 0.f, 0.f};
        if (aok && kr < K) {
          int e  = tabLDS[kr];
          int c0 = e & 1023;
          int sx = (e >> 10) & 31;
          int rx = e >> 15;
          int ih = ihB + rx, iw = iwB + sx;
          if ((unsigned)ih < (unsigned)Hin && (unsigned)iw < (unsigned)Win) {
            const float4* p = (const float4*)(in + aBase + ih * WinC + iw * Cin + c0);
            float4 lo = p[0], hi = p[1];
            buf[0] = lo.x; buf[1] = lo.y; buf[2] = lo.z; buf[3] = lo.w;
            buf[4] = hi.x; buf[5] = hi.y; buf[6] = hi.z; buf[7] = hi.w;
          }
        }
#pragma unroll
        for (int j = 0; j < 8; ++j) a[h * 8 + j] = (__bf16)buf[j];
      }
    } else {
      // scalar path (stem conv, Cin==3)
#pragma unroll
      for (int i = 0; i < 16; ++i) {
        int kk = k0 + akb + ((i < 8) ? i : (i + 8));
        float av = 0.f;
        if (aok && kk < K) {
          int e  = tabLDS[kk];
          int c  = e & 1023;
          int sx = (e >> 10) & 31;
          int rx = e >> 15;
          int ih = ihB + rx, iw = iwB + sx;
          if ((unsigned)ih < (unsigned)Hin && (unsigned)iw < (unsigned)Win)
            av = in[aBase + ih * WinC + iw * Cin + c];
        }
        a[i] = (__bf16)av;
      }
    }
  };

  v8f acc0 = {}, acc1 = {};
  for (int k0 = 0; k0 < Kp; k0 += 64) {
    __builtin_prefetch(wrow + k0 + 64, 0, 1);  // global_prefetch_b8 (L2-resident weights)
    {
      v16bf a0;
      loadA(k0, a0);
      const v8bf* wp = (const v8bf*)(wrow + k0);
      v8bf blo = wp[0], bhi = wp[2];
      v16bf b0 = __builtin_shufflevector(blo, bhi, 0, 1, 2, 3, 4, 5, 6, 7,
                                         8, 9, 10, 11, 12, 13, 14, 15);
      acc0 = __builtin_amdgcn_wmma_f32_16x16x32_bf16(
          false, a0, false, b0, (short)0, acc0, false, false);
    }
    if (k0 + 32 < Kp) {
      v16bf a1;
      loadA(k0 + 32, a1);
      const v8bf* wp = (const v8bf*)(wrow + k0 + 32);
      v8bf blo = wp[0], bhi = wp[2];
      v16bf b1 = __builtin_shufflevector(blo, bhi, 0, 1, 2, 3, 4, 5, 6, 7,
                                         8, 9, 10, 11, 12, 13, 14, 15);
      acc1 = __builtin_amdgcn_wmma_f32_16x16x32_bf16(
          false, a1, false, b1, (short)0, acc1, false, false);
    }
  }
#pragma unroll
  for (int r = 0; r < 8; ++r) acc0[r] += acc1[r];

  // store D
  const int cn    = nt * 16 + (lane & 15);
  const int rbase = (lane >> 4) * 8;
  if (cn < Cout) {
#pragma unroll
    for (int r = 0; r < 8; ++r) {
      int gm = mt * 16 + rbase + r;
      if (gm < M) {
        float v = acc0[r];
        if (mask)  v *= mask[gm];
        if (bias)  v += bias[cn];
        int oidx = gm * Cout + cn;
        if (resid) v += resid[oidx];
        out[oidx] = v;
      }
    }
  }
}

// ---------------------------------------------------------------------------
static inline long cdiv(long a, long b) { return (a + b - 1) / b; }

extern "C" void kernel_launch(void* const* d_in, const int* in_sizes, int n_in,
                              void* d_out, int out_size, void* d_ws, size_t ws_size,
                              hipStream_t stream) {
  (void)in_sizes; (void)n_in; (void)out_size; (void)ws_size;
  const int B = 128;

  // ---- inputs (setup_inputs dict order, params in insertion order) ----
  int t = 0;
  const float* x   = (const float*)d_in[t++];   // [128,95,95,3]
  const float* m95 = (const float*)d_in[t++];   // [128,95,95,1]
  const float* w0  = (const float*)d_in[t++];   // [3,3,3,16]

  struct BP { const float *g1, *b1, *w1, *g2, *b2, *w2, *wr; int cin, cout, s; };
  BP bp[8];
  const int gcfg[4][3] = {{16, 2, 1}, {32, 2, 2}, {48, 2, 2}, {96, 2, 2}};
  {
    int nIn = 16, bi = 0;
    for (int g = 0; g < 4; ++g)
      for (int rep = 0; rep < 2; ++rep, ++bi) {
        int n = gcfg[g][0];
        int s = (rep == 0) ? gcfg[g][2] : 1;
        bp[bi].g1 = (const float*)d_in[t++];
        bp[bi].b1 = (const float*)d_in[t++];
        bp[bi].w1 = (const float*)d_in[t++];
        bp[bi].g2 = (const float*)d_in[t++];
        bp[bi].b2 = (const float*)d_in[t++];
        bp[bi].w2 = (const float*)d_in[t++];
        bp[bi].wr = (s > 1 || nIn != n) ? (const float*)d_in[t++] : nullptr;
        bp[bi].cin = nIn; bp[bi].cout = n; bp[bi].s = s;
        nIn = n;
      }
  }
  const float* bnf_g = (const float*)d_in[t++];
  const float* bnf_b = (const float*)d_in[t++];
  const float* w5    = (const float*)d_in[t++];  // [5,5,96,128]
  const float* bn5_g = (const float*)d_in[t++];
  const float* bn5_b = (const float*)d_in[t++];
  const float* lin_w = (const float*)d_in[t++];  // [128,3755]
  const float* lin_b = (const float*)d_in[t++];

  // ---- workspace carving ----
  uintptr_t cur = (uintptr_t)d_ws;
  auto alloc = [&](size_t bytes) -> void* {
    uintptr_t p = (cur + 255) & ~(uintptr_t)255;
    cur = p + bytes;
    return (void*)p;
  };
  float* bufBig = (float*)alloc((size_t)B * 95 * 95 * 16 * 4);  // conv0 out
  float* R[4];
  for (int i = 0; i < 4; ++i) R[i] = (float*)alloc((size_t)B * 47 * 47 * 16 * 4);
  float* m47  = (float*)alloc((size_t)B * 47 * 47 * 4);
  float* m23  = (float*)alloc((size_t)B * 23 * 23 * 4);
  float* m11  = (float*)alloc((size_t)B * 11 * 11 * 4);
  float* m5s  = (float*)alloc((size_t)B * 5 * 5 * 4);
  float* m1s  = (float*)alloc((size_t)B * 4);
  float* stats = (float*)alloc(257 * 4);          // [2*Cmax sums | cnt]
  __bf16* wpool = (__bf16*)alloc(2000000ull * 2); // bf16 transposed weight pool
  __bf16* wcur  = wpool;
  int* tabpool = (int*)alloc(20000 * 4);          // im2col table pool
  int* tcur    = tabpool;

  // pack [K,Cout] f32 -> [Coutp,Kp] bf16 (transposed, zero padded)
  auto packT = [&](const float* src, int K, int Cout) -> const __bf16* {
    int Kp = (K + 31) & ~31;
    int Coutp = (Cout + 15) & ~15;
    int np = Coutp * Kp;
    __bf16* dst = wcur; wcur += np;
    k_pack_wT<<<dim3((unsigned)cdiv(np, 256)), dim3(256), 0, stream>>>(
        src, dst, K, Cout, Kp, np);
    return dst;
  };

  auto run_conv = [&](const float* in, const __bf16* wt, const float* mask,
                      const float* resid, const float* bias, float* out,
                      int Hin, int Win, int Cin, int Cout, int Hout, int Wout,
                      int kh, int kw, int stride, int pad) {
    int K = kh * kw * Cin;
    int Kp = (K + 31) & ~31;
    int* tab = tcur; tcur += K;
    k_build_tab<<<dim3((unsigned)cdiv(K, 128)), dim3(128), 0, stream>>>(tab, kw, Cin, K);
    int M = B * Hout * Wout;
    int fastA = ((Cin & 7) == 0) ? 1 : 0;
    dim3 grid((unsigned)cdiv(M, 16), (unsigned)cdiv(Cout, 16));
    k_conv_wmma<<<grid, dim3(32), 0, stream>>>(in, wt, tab, mask, resid, bias, out,
                                               M, Hin, Win, Cin, Cout,
                                               Wout, Hout * Wout, stride, pad,
                                               Kp, K, fastA);
  };

  auto run_bn = [&](const float* xin, float* xout, const float* mm,
                    int nsites, int C, const float* g, const float* b) {
    hipMemsetAsync(stats, 0, (2 * C + 1) * sizeof(float), stream);
    float* cnt = stats + 2 * C;
    int n = nsites * C;
    int cblocks = (int)cdiv(nsites, 256); if (cblocks > 1024) cblocks = 1024;
    k_mask_count<<<dim3((unsigned)cblocks), dim3(256), 0, stream>>>(mm, cnt, nsites);
    int rblocks = (int)cdiv(n, 256); if (rblocks > 2048) rblocks = 2048;
    k_bn_reduce<<<dim3((unsigned)rblocks), dim3(256), 2 * C * sizeof(float), stream>>>(
        xin, stats, C, n);
    k_bn_apply<<<dim3((unsigned)cdiv(n, 256)), dim3(256), 0, stream>>>(
        xin, mm, stats, cnt, g, b, xout, n, C);
  };

  // ---- stem: submanifold conv 3->16 @95, masked max-pool 3x3/2 -> 47 ----
  const __bf16* w0b = packT(w0, 27, 16);
  run_conv(x, w0b, m95, nullptr, nullptr, bufBig, 95, 95, 3, 16, 95, 95, 3, 3, 1, 1);
  {
    int n = B * 47 * 47;
    k_pool_mask<<<dim3((unsigned)cdiv(n, 256)), dim3(256), 0, stream>>>(
        m95, m47, B, 95, 95, 47, 47, 3, 2);
    int ne = n * 16;
    k_maxpool_active<<<dim3((unsigned)cdiv(ne, 256)), dim3(256), 0, stream>>>(
        bufBig, m95, m47, R[0], B, 95, 95, 16, 47, 47, 3, 2);
  }

  // ---- residual blocks ----
  int xi = 0;       // region index holding block input X
  int din = 47;
  const float* mIn = m47;
  float* mlev[3] = {m23, m11, m5s};
  int mnext = 0;

  for (int blk = 0; blk < 8; ++blk) {
    BP& p = bp[blk];
    int others[3], oi = 0;
    for (int j = 0; j < 4; ++j) if (j != xi) others[oi++] = j;
    float* T1 = R[others[0]];
    float* T2 = R[others[1]];
    float* T3 = R[others[2]];
    const float* X = R[xi];

    int dout = din;
    const float* mOut = mIn;
    if (p.s == 2) {
      dout = (din - 3) / 2 + 1;
      float* mo = mlev[mnext++];
      int n = B * dout * dout;
      k_pool_mask<<<dim3((unsigned)cdiv(n, 256)), dim3(256), 0, stream>>>(
          mIn, mo, B, din, din, dout, dout, 3, 2);
      mOut = mo;
    }

    const __bf16* w1b = packT(p.w1, 9 * p.cin, p.cout);
    const __bf16* w2b = packT(p.w2, 9 * p.cout, p.cout);

    // pre-activation BN-ReLU
    run_bn(X, T1, mIn, B * din * din, p.cin, p.g1, p.b1);
    // conv1 (submanifold SAME, or strided VALID)
    if (p.s == 1)
      run_conv(T1, w1b, mOut, nullptr, nullptr, T2, din, din, p.cin, p.cout,
               din, din, 3, 3, 1, 1);
    else
      run_conv(T1, w1b, mOut, nullptr, nullptr, T2, din, din, p.cin, p.cout,
               dout, dout, 3, 3, 2, 0);
    // BN-ReLU
    run_bn(T2, T1, mOut, B * dout * dout, p.cout, p.g2, p.b2);
    // residual branch
    const float* resid;
    if (p.wr) {
      if (p.s == 2) {
        const __bf16* wrb = packT(p.wr, 9 * p.cin, p.cout);
        run_conv(X, wrb, mOut, nullptr, nullptr, T3, din, din, p.cin, p.cout,
                 dout, dout, 3, 3, 2, 0);
      } else {
        const __bf16* wrb = packT(p.wr, p.cin, p.cout);
        run_conv(X, wrb, mOut, nullptr, nullptr, T3, din, din, p.cin, p.cout,
                 din, din, 1, 1, 1, 0);
      }
      resid = T3;
    } else {
      resid = X;
    }
    // conv2 (submanifold SAME) fused with mask*out + residual
    run_conv(T1, w2b, mOut, resid, nullptr, T2, dout, dout, p.cout, p.cout,
             dout, dout, 3, 3, 1, 1);

    xi = others[1];
    din = dout;
    mIn = mOut;
  }

  // ---- head ----
  {
    const float* X = R[xi];
    float* T1 = R[(xi + 1) & 3];
    float* T2 = R[(xi + 2) & 3];

    run_bn(X, T1, m5s, B * 5 * 5, 96, bnf_g, bnf_b);

    int n = B;
    k_pool_mask<<<dim3((unsigned)cdiv(n, 256)), dim3(256), 0, stream>>>(
        m5s, m1s, B, 5, 5, 1, 1, 5, 1);

    const __bf16* w5b = packT(w5, 25 * 96, 128);
    run_conv(T1, w5b, m1s, nullptr, nullptr, T2, 5, 5, 96, 128, 1, 1, 5, 5, 1, 0);

    run_bn(T2, T1, m1s, B, 128, bn5_g, bn5_b);

    const __bf16* lwb = packT(lin_w, 128, 3755);
    run_conv(T1, lwb, nullptr, nullptr, lin_b, (float*)d_out,
             1, 1, 128, 3755, 1, 1, 1, 1, 1, 0);
  }
}
